// GCNFusion_34102040330885
// MI455X (gfx1250) — matrix operand order repaired and checked by reading, not compiled, not run on hardware
//
#include <hip/hip_runtime.h>
#include <cstdint>
#include <cstddef>

// ---------------------------------------------------------------------------
// Problem constants (from reference)
// ---------------------------------------------------------------------------
#define N_NODES 100000
#define N_EDGES 1280000
#define IN_CH   500
#define TEXT_DIM 384
#define HIDDEN  64
#define OUT_CH  7

typedef __attribute__((ext_vector_type(16))) __bf16 v16bf;
typedef __attribute__((ext_vector_type(8)))  float  v8f;

// ---------------------------------------------------------------------------
// Utility kernels
// ---------------------------------------------------------------------------
__global__ void k_zero(float* __restrict__ p, long n) {
    long i = (long)blockIdx.x * blockDim.x + threadIdx.x;
    if (i < n) p[i] = 0.0f;
}

// in-degree count via float atomics (deg accumulated in dinv buffer)
__global__ void k_deg(const long long* __restrict__ eidx, float* __restrict__ deg, int E) {
    int i = blockIdx.x * blockDim.x + threadIdx.x;
    if (i < E) {
        int d = (int)eidx[E + i];           // dst row of edge_index
        atomicAdd(&deg[d], 1.0f);
    }
}

// deg -> (deg+1)^-0.5   (self-loop included analytically, like the reference)
__global__ void k_dinv(float* __restrict__ p, int n) {
    int i = blockIdx.x * blockDim.x + threadIdx.x;
    if (i < n) p[i] = rsqrtf(p[i] + 1.0f);
}

// ---------------------------------------------------------------------------
// Split f32 weights [K x Ns] into bf16 hi/lo fragments, PRE-SWIZZLED into
// WMMA B-matrix lane order (32x16 16-bit B layout, ISA 7.12.2):
//   lanes 0-15: col = lane,    K = kstep*32 + e (e<8), kstep*32 + 16 + (e-8) (e>=8)
//   lanes 16-31: col = lane-16, K = kstep*32 + 8 + e (e<8), kstep*32 + 24 + (e-8) (e>=8)
// Storage:  frag[((kstep*NT + t)*32 + lane)*16 + e]
// Zero-padded in both K (to mult. of 32) and cols (to 16*NT), so the GEMM
// needs no bounds checks on B at all.
// ---------------------------------------------------------------------------
__global__ void k_cvt_w_swz(const float* __restrict__ w, __bf16* __restrict__ hi,
                            __bf16* __restrict__ lo, int K, int Ns, int NT, int Ksteps) {
    int i = blockIdx.x * blockDim.x + threadIdx.x;
    int total = Ksteps * NT * 512;            // 32 lanes * 16 elems
    if (i < total) {
        int e     = i & 15;
        int lane  = (i >> 4) & 31;
        int t     = (i >> 9) % NT;
        int kstep = i / (512 * NT);
        int k   = kstep * 32 + ((lane >> 4) ? 8 : 0) + e + ((e & 8) ? 8 : 0);
        int col = t * 16 + (lane & 15);
        float v = (k < K && col < Ns) ? w[(size_t)k * Ns + col] : 0.0f;
        __bf16 h = (__bf16)v;                 // RNE
        hi[i] = h;
        lo[i] = (__bf16)(v - (float)h);       // residual
    }
}

// ---------------------------------------------------------------------------
// Fragment load helpers
// ---------------------------------------------------------------------------
__device__ __forceinline__ void split4(const float4 f, v16bf& hi, v16bf& lo, int base) {
    const float v[4] = {f.x, f.y, f.z, f.w};
    #pragma unroll
    for (int j = 0; j < 4; ++j) {
        __bf16 h = (__bf16)v[j];
        hi[base + j] = h;
        lo[base + j] = (__bf16)(v[j] - (float)h);
    }
}

// A fragment (16-bit A 16x32 layout): lane reads its row at two contiguous
// 8-float runs: [kb, kb+8) and [kb+16, kb+24), kb = kk + half*8.
template<bool MASK>
__device__ __forceinline__ void load_a_frag(const float* __restrict__ ap, int kb, int K,
                                            v16bf& ahi, v16bf& alo) {
    if (!MASK) {
        float4 a0 = *(const float4*)(ap + kb);
        float4 a1 = *(const float4*)(ap + kb + 4);
        float4 a2 = *(const float4*)(ap + kb + 16);
        float4 a3 = *(const float4*)(ap + kb + 20);
        split4(a0, ahi, alo, 0);
        split4(a1, ahi, alo, 4);
        split4(a2, ahi, alo, 8);
        split4(a3, ahi, alo, 12);
    } else {
        #pragma unroll
        for (int e = 0; e < 16; ++e) {
            int k = kb + e + ((e & 8) ? 8 : 0);     // e<8: kb+e ; e>=8: kb+16+(e-8)
            float v = (k < K) ? ap[k] : 0.0f;
            __bf16 h = (__bf16)v;
            ahi[e] = h;
            alo[e] = (__bf16)(v - (float)h);
        }
    }
}

// ---------------------------------------------------------------------------
// WMMA GEMM:  Out[M x 16*NT] (+bias) = A[M x K] * B  with B pre-swizzled.
// A f32, split to bf16 hi/lo on the fly.  3x bf16 WMMA emulates ~f32:
//     D += Ahi*Bhi + Alo*Bhi + Ahi*Blo
// One wave computes a 16 x (16*NT) tile.  8 waves / 256-thread block.
// Main K loop is bounds-check free; single masked tail step handles K%32.
// ---------------------------------------------------------------------------
template<int NT>
__global__ __launch_bounds__(256) void k_gemm_bf16x3(
    const float* __restrict__ A, int lda, int K,
    const __bf16* __restrict__ Bhi, const __bf16* __restrict__ Blo,
    const float* __restrict__ bias, int has_bias,
    float* __restrict__ Out, int ldo, int col0, int ncols_store, int Mtiles)
{
    int wave = blockIdx.x * (blockDim.x >> 5) + (threadIdx.x >> 5);
    if (wave >= Mtiles) return;             // wave-uniform guard (EXEC all-1s for WMMA)

    const int lane = threadIdx.x & 31;
    const int half = lane >> 4;             // 0: lanes 0-15, 1: lanes 16-31
    const int l15  = lane & 15;
    const float* __restrict__ ap = A + (size_t)(wave * 16 + l15) * lda;

    v8f acc[NT] = {};

    const int Kmain = K & ~31;
    for (int kk = 0; kk < Kmain; kk += 32) {
        v16bf ahi, alo;
        load_a_frag<false>(ap, kk + half * 8, K, ahi, alo);
        const int kstep = kk >> 5;
        #pragma unroll
        for (int t = 0; t < NT; ++t) {
            size_t fo = ((size_t)(kstep * NT + t) * 32 + lane) * 16;
            v16bf bhi = *(const v16bf*)(Bhi + fo);
            v16bf blo = *(const v16bf*)(Blo + fo);
            acc[t] = __builtin_amdgcn_wmma_f32_16x16x32_bf16(
                         false, ahi, false, bhi, (short)0, acc[t], false, false);
            acc[t] = __builtin_amdgcn_wmma_f32_16x16x32_bf16(
                         false, alo, false, bhi, (short)0, acc[t], false, false);
            acc[t] = __builtin_amdgcn_wmma_f32_16x16x32_bf16(
                         false, ahi, false, blo, (short)0, acc[t], false, false);
        }
    }
    if (Kmain < K) {                        // one masked tail step (B is zero-padded)
        v16bf ahi, alo;
        load_a_frag<true>(ap, Kmain + half * 8, K, ahi, alo);
        const int kstep = Kmain >> 5;
        #pragma unroll
        for (int t = 0; t < NT; ++t) {
            size_t fo = ((size_t)(kstep * NT + t) * 32 + lane) * 16;
            v16bf bhi = *(const v16bf*)(Bhi + fo);
            v16bf blo = *(const v16bf*)(Blo + fo);
            acc[t] = __builtin_amdgcn_wmma_f32_16x16x32_bf16(
                         false, ahi, false, bhi, (short)0, acc[t], false, false);
            acc[t] = __builtin_amdgcn_wmma_f32_16x16x32_bf16(
                         false, alo, false, bhi, (short)0, acc[t], false, false);
            acc[t] = __builtin_amdgcn_wmma_f32_16x16x32_bf16(
                         false, ahi, false, blo, (short)0, acc[t], false, false);
        }
    }

    // ---- store D (32-bit C/D 16x16 layout: VGPR r -> M = r + half*8) ----
    #pragma unroll
    for (int t = 0; t < NT; ++t) {
        const int col = t * 16 + l15;
        if (col < ncols_store) {
            float bv = has_bias ? bias[col] : 0.0f;
            #pragma unroll
            for (int r = 0; r < 8; ++r) {
                int m = wave * 16 + half * 8 + r;
                Out[(size_t)m * ldo + col0 + col] = acc[t][r] + bv;
            }
        }
    }
}

// ---------------------------------------------------------------------------
// Edge scatter, layer 1: 64 channels.  One work-item = (edge, 4-channel quad).
// ---------------------------------------------------------------------------
__global__ void k_scatter64(const long long* __restrict__ eidx,
                            const float* __restrict__ dinv,
                            const float* __restrict__ hlin,
                            float* __restrict__ agg, int E) {
    long idx = (long)blockIdx.x * blockDim.x + threadIdx.x;
    if (idx >= (long)E * 16) return;
    int e = (int)(idx >> 4);
    int q = (int)(idx & 15);
    int s = (int)eidx[e];
    int d = (int)eidx[E + e];
    float coef = dinv[s] * dinv[d];
    const float4 v = *(const float4*)(hlin + (size_t)s * 64 + q * 4);
    float* base = agg + (size_t)d * 64 + q * 4;
    atomicAdd(base + 0, v.x * coef);
    atomicAdd(base + 1, v.y * coef);
    atomicAdd(base + 2, v.z * coef);
    atomicAdd(base + 3, v.w * coef);
}

// h1 = relu(agg + hlin * dinv^2 + b1), written in place over agg
__global__ void k_combine1(float* __restrict__ agg, const float* __restrict__ hlin,
                           const float* __restrict__ dinv, const float* __restrict__ b1,
                           int N) {
    long i = (long)blockIdx.x * blockDim.x + threadIdx.x;
    if (i < (long)N * 64) {
        int node = (int)(i >> 6);
        int c    = (int)(i & 63);
        float di = dinv[node];
        float v = agg[i] + hlin[i] * di * di + b1[c];
        agg[i] = v > 0.0f ? v : 0.0f;
    }
}

// Edge scatter, layer 2: 7 channels, one thread per edge, accumulate into d_out
__global__ void k_scatter7(const long long* __restrict__ eidx,
                           const float* __restrict__ dinv,
                           const float* __restrict__ h2,   // stride 8, cols 0..6 valid
                           float* __restrict__ out, int E) {
    int e = blockIdx.x * blockDim.x + threadIdx.x;
    if (e < E) {
        int s = (int)eidx[e];
        int d = (int)eidx[E + e];
        float coef = dinv[s] * dinv[d];
        const float* hp = h2 + (size_t)s * 8;
        float* op = out + (size_t)d * 7;
        #pragma unroll
        for (int c = 0; c < 7; ++c) atomicAdd(op + c, hp[c] * coef);
    }
}

// out += h2 * dinv^2 + b2   (self-loop term + bias)
__global__ void k_final(float* __restrict__ out, const float* __restrict__ h2,
                        const float* __restrict__ dinv, const float* __restrict__ b2,
                        int N) {
    long i = (long)blockIdx.x * blockDim.x + threadIdx.x;
    if (i < (long)N * 7) {
        int node = (int)(i / 7);
        int c    = (int)(i % 7);
        float di = dinv[node];
        out[i] += h2[(size_t)node * 8 + c] * di * di + b2[c];
    }
}

// ---------------------------------------------------------------------------
// Host launcher
// ---------------------------------------------------------------------------
static inline size_t align_up(size_t x, size_t a) { return (x + a - 1) & ~(a - 1); }

extern "C" void kernel_launch(void* const* d_in, const int* in_sizes, int n_in,
                              void* d_out, int out_size, void* d_ws, size_t ws_size,
                              hipStream_t stream) {
    (void)in_sizes; (void)n_in; (void)out_size; (void)ws_size;

    const float*     x      = (const float*)d_in[0];
    const float*     text_x = (const float*)d_in[1];
    const long long* eidx   = (const long long*)d_in[2];   // [2, E] int64: src then dst
    const float*     w_feat = (const float*)d_in[3];
    const float*     b_feat = (const float*)d_in[4];
    const float*     w_text = (const float*)d_in[5];
    const float*     b_text = (const float*)d_in[6];
    const float*     w1     = (const float*)d_in[7];
    const float*     b1     = (const float*)d_in[8];
    const float*     w2     = (const float*)d_in[9];
    const float*     b2     = (const float*)d_in[10];
    float*           out    = (float*)d_out;

    // swizzled-fragment element counts:  Ksteps * NT * 32 lanes * 16 elems
    const int KS_F = (IN_CH + 31) / 32;     // 16
    const int KS_T = (TEXT_DIM + 31) / 32;  // 12
    const int KS_1 = 128 / 32;              // 4
    const int KS_2 = 64 / 32;               // 2
    const int NF = KS_F * 4 * 512;          // 32768
    const int NTx = KS_T * 4 * 512;         // 24576
    const int N1 = KS_1 * 4 * 512;          // 8192
    const int N2 = KS_2 * 1 * 512;          // 1024

    // ---- workspace layout ----
    char* ws = (char*)d_ws;
    size_t off = 0;
    float* dinv  = (float*)(ws + off); off = align_up(off + (size_t)N_NODES * 4, 256);
    float* h0    = (float*)(ws + off); off = align_up(off + (size_t)N_NODES * 128 * 4, 256);
    float* hlin1 = (float*)(ws + off); off = align_up(off + (size_t)N_NODES * 64 * 4, 256);
    float* hlin2 = (float*)(ws + off); off = align_up(off + (size_t)N_NODES * 8 * 4, 256);
    __bf16* wf_hi = (__bf16*)(ws + off); off = align_up(off + (size_t)NF * 2, 256);
    __bf16* wf_lo = (__bf16*)(ws + off); off = align_up(off + (size_t)NF * 2, 256);
    __bf16* wt_hi = (__bf16*)(ws + off); off = align_up(off + (size_t)NTx * 2, 256);
    __bf16* wt_lo = (__bf16*)(ws + off); off = align_up(off + (size_t)NTx * 2, 256);
    __bf16* w1_hi = (__bf16*)(ws + off); off = align_up(off + (size_t)N1 * 2, 256);
    __bf16* w1_lo = (__bf16*)(ws + off); off = align_up(off + (size_t)N1 * 2, 256);
    __bf16* w2_hi = (__bf16*)(ws + off); off = align_up(off + (size_t)N2 * 2, 256);
    __bf16* w2_lo = (__bf16*)(ws + off); off = align_up(off + (size_t)N2 * 2, 256);
    float* agg1 = h0;   // alias: h0 dead after GEMM2; agg1/h1 live after
    float* h1   = agg1; // combine1 writes h1 in place

    const int T = 256;
    const int Mtiles = N_NODES / 16;                  // 6250
    const int gemm_blocks = (Mtiles + 7) / 8;         // 8 waves per block

    // 1) dinv = rsqrt(in_degree + 1)
    k_zero<<<(N_NODES + T - 1) / T, T, 0, stream>>>(dinv, N_NODES);
    k_deg<<<(N_EDGES + T - 1) / T, T, 0, stream>>>(eidx, dinv, N_EDGES);
    k_dinv<<<(N_NODES + T - 1) / T, T, 0, stream>>>(dinv, N_NODES);

    // 2) split + swizzle weights into WMMA B fragments (bf16 hi/lo)
    k_cvt_w_swz<<<(NF + T - 1) / T, T, 0, stream>>>(w_feat, wf_hi, wf_lo, IN_CH, 64, 4, KS_F);
    k_cvt_w_swz<<<(NTx + T - 1) / T, T, 0, stream>>>(w_text, wt_hi, wt_lo, TEXT_DIM, 64, 4, KS_T);
    k_cvt_w_swz<<<(N1 + T - 1) / T, T, 0, stream>>>(w1, w1_hi, w1_lo, 128, 64, 4, KS_1);
    k_cvt_w_swz<<<(N2 + T - 1) / T, T, 0, stream>>>(w2, w2_hi, w2_lo, 64, OUT_CH, 1, KS_2);

    // 3) h0[:,0:64]  = x @ w_feat + b_feat ;  h0[:,64:128] = text_x @ w_text + b_text
    k_gemm_bf16x3<4><<<gemm_blocks, T, 0, stream>>>(
        x, IN_CH, IN_CH, wf_hi, wf_lo, b_feat, 1, h0, 128, 0, 64, Mtiles);
    k_gemm_bf16x3<4><<<gemm_blocks, T, 0, stream>>>(
        text_x, TEXT_DIM, TEXT_DIM, wt_hi, wt_lo, b_text, 1, h0, 128, 64, 64, Mtiles);

    // 4) hlin1 = h0 @ w1  (bias deferred to post-aggregation)
    k_gemm_bf16x3<4><<<gemm_blocks, T, 0, stream>>>(
        h0, 128, 128, w1_hi, w1_lo, nullptr, 0, hlin1, 64, 0, 64, Mtiles);

    // 5) layer-1 aggregation + self-loop + bias + relu
    k_zero<<<(int)(((long)N_NODES * 64 + T - 1) / T), T, 0, stream>>>(agg1, (long)N_NODES * 64);
    k_scatter64<<<(int)(((long)N_EDGES * 16 + T - 1) / T), T, 0, stream>>>(
        eidx, dinv, hlin1, agg1, N_EDGES);
    k_combine1<<<(int)(((long)N_NODES * 64 + T - 1) / T), T, 0, stream>>>(
        agg1, hlin1, dinv, b1, N_NODES);

    // 6) hlin2 = h1 @ w2 (padded to 16 cols; stored stride 8)
    k_gemm_bf16x3<1><<<gemm_blocks, T, 0, stream>>>(
        h1, 64, 64, w2_hi, w2_lo, nullptr, 0, hlin2, 8, 0, 8, Mtiles);

    // 7) layer-2 aggregation straight into d_out, then self-loop + bias
    k_zero<<<(int)(((long)N_NODES * 7 + T - 1) / T), T, 0, stream>>>(out, (long)N_NODES * 7);
    k_scatter7<<<(N_EDGES + T - 1) / T, T, 0, stream>>>(eidx, dinv, hlin2, out, N_EDGES);
    k_final<<<(int)(((long)N_NODES * 7 + T - 1) / T), T, 0, stream>>>(
        out, hlin2, dinv, b2, N_NODES);
}
